// RGATLayer_54528904790451
// MI455X (gfx1250) — matrix operand order repaired and checked by reading, not compiled, not run on hardware
//
#include <hip/hip_runtime.h>
#include <hip/hip_bf16.h>
#include <stdint.h>

// Problem constants (match reference)
#define NN   100000   // nodes
#define EE   500000   // edges per relation
#define INF_ 128      // input features
#define HH   4        // heads
#define DD   64       // per-head out dim
#define HD   256      // HH*DD
#define NEG_SLOPE 0.2f

typedef __attribute__((ext_vector_type(16))) __bf16 v16bf;
typedef __attribute__((ext_vector_type(8)))  __bf16 v8bf;
typedef __attribute__((ext_vector_type(4)))  __bf16 v4bf;
typedef __attribute__((ext_vector_type(8)))  float  v8f;

// ---------------------------------------------------------------------------
// float atomic-max via signed/unsigned integer atomics (no CAS loop).
// float ordering == int ordering for x>=0, reversed-uint ordering for x<0.
// mbuf is initialized to -inf (0xFF800000): identity for both ops.
__device__ inline void atomicMaxF32(float* addr, float val) {
  if (!(val == val)) return;                    // skip NaN
  if (val >= 0.0f) atomicMax((int*)addr, __float_as_int(val));
  else             atomicMin((unsigned int*)addr, __float_as_uint(val));
}

// ---------------------------------------------------------------------------
// x [N][128] fp32 -> bf16, 4 elems/thread
__global__ __launch_bounds__(256) void k_cvt_x(const float* __restrict__ x,
                                               __bf16* __restrict__ xb) {
  size_t i = ((size_t)blockIdx.x * 256 + threadIdx.x) * 4;
  float4 v = *(const float4*)(x + i);
  v4bf o = { (__bf16)v.x, (__bf16)v.y, (__bf16)v.z, (__bf16)v.w };
  *(v4bf*)(xb + i) = o;
}

// W [128][256] fp32 row-major -> wb [256][128] bf16 (transposed: contiguous K per col)
__global__ __launch_bounds__(256) void k_cvt_w(const float* __restrict__ W,
                                               __bf16* __restrict__ wb) {
  int k = blockIdx.x;          // 0..127
  int n = threadIdx.x;         // 0..255
  wb[n * INF_ + k] = (__bf16)W[k * HD + n];
}

// out[n][c] = b0[c]+b1[c]+b2[c]
__global__ __launch_bounds__(256) void k_init_out(const float* __restrict__ b0,
                                                  const float* __restrict__ b1,
                                                  const float* __restrict__ b2,
                                                  float* __restrict__ out) {
  size_t i = (size_t)blockIdx.x * 256 + threadIdx.x;   // grid = NN blocks
  int c = (int)(i & (HD - 1));
  out[i] = b0[c] + b1[c] + b2[c];
}

// mbuf = -inf, sbuf = 0  (per relation, re-run each replay)
__global__ __launch_bounds__(256) void k_init_ms(float* __restrict__ mbuf,
                                                 float* __restrict__ sbuf) {
  int i = blockIdx.x * 256 + threadIdx.x;
  if (i < NN * HH) { mbuf[i] = -INFINITY; sbuf[i] = 0.0f; }
}

// ---------------------------------------------------------------------------
// WMMA GEMM: feat[32-row slab][256] = x_bf16 @ W_bf16, fp32 accumulate.
// Block = 256 thr = 8 waves. Each wave owns a 32x32 output patch:
// row tiles {0,16} x col tiles {32w, 32w+16}. A slab (32x128 bf16 = 8KB) in LDS.
// Each A and B fragment feeds two WMMAs -> 16 v_wmma per block body, and B
// L2 traffic per output element is halved vs a 16-row slab.
__global__ __launch_bounds__(256) void k_gemm_wmma(const __bf16* __restrict__ xb,
                                                   const __bf16* __restrict__ wb,
                                                   float* __restrict__ feat) {
  __shared__ __bf16 lds_a[32 * INF_];          // 8KB
  const int rowbase = blockIdx.x * 32;         // 3125 blocks, N divisible by 32
  const int tid = threadIdx.x;

  // cooperative 8KB copy: 256 threads x 2 x 16B
  {
    const uint4* s = (const uint4*)(xb + (size_t)rowbase * INF_);
    uint4* d = (uint4*)lds_a;
    d[tid]       = s[tid];
    d[tid + 256] = s[tid + 256];
  }
  __syncthreads();

  const int wave = tid >> 5;
  const int lane = tid & 31;
  const int m    = lane & 15;      // A row / B col / C col within tile
  const int half = lane >> 4;      // lane-group select (wave32)
  const int c0 = wave * 32;        // first col tile base
  const int c1 = wave * 32 + 16;   // second col tile base

  v8f acc00 = {}, acc01 = {};      // row tile 0 x col tiles {0,1}
  v8f acc10 = {}, acc11 = {};      // row tile 1 x col tiles {0,1}

#pragma unroll
  for (int kb = 0; kb < INF_; kb += 32) {
    // A fragments (16x32 bf16): lane holds contiguous K = kb+half*16+0..15
    v16bf a0, a1;
    {
      const v8bf lo0 = *(const v8bf*)(lds_a + m * INF_ + kb + half * 16);
      const v8bf hi0 = *(const v8bf*)(lds_a + m * INF_ + kb + half * 16 + 8);
      const v8bf lo1 = *(const v8bf*)(lds_a + (m + 16) * INF_ + kb + half * 16);
      const v8bf hi1 = *(const v8bf*)(lds_a + (m + 16) * INF_ + kb + half * 16 + 8);
#pragma unroll
      for (int i = 0; i < 8; ++i) {
        a0[i] = lo0[i]; a0[i + 8] = hi0[i];
        a1[i] = lo1[i]; a1[i + 8] = hi1[i];
      }
    }
    // B fragments (32x16 bf16): lane holds contiguous K = kb+half*16+0..15 of its column
    const v16bf b0 = *(const v16bf*)(wb + (size_t)(c0 + m) * INF_ + kb + half * 16);
    const v16bf b1 = *(const v16bf*)(wb + (size_t)(c1 + m) * INF_ + kb + half * 16);

    acc00 = __builtin_amdgcn_wmma_f32_16x16x32_bf16(false, a0, false, b0,
                                                    (short)0, acc00, false, false);
    acc01 = __builtin_amdgcn_wmma_f32_16x16x32_bf16(false, a0, false, b1,
                                                    (short)0, acc01, false, false);
    acc10 = __builtin_amdgcn_wmma_f32_16x16x32_bf16(false, a1, false, b0,
                                                    (short)0, acc10, false, false);
    acc11 = __builtin_amdgcn_wmma_f32_16x16x32_bf16(false, a1, false, b1,
                                                    (short)0, acc11, false, false);
  }

  // C layout: VGPR j -> row j + 8*half (+16 for row tile 1), col = m
  float* frow = feat + (size_t)rowbase * HD;
#pragma unroll
  for (int j = 0; j < 8; ++j) {
    const int r0 = j + 8 * half;
    const int r1 = r0 + 16;
    frow[(size_t)r0 * HD + c0 + m] = acc00[j];
    frow[(size_t)r0 * HD + c1 + m] = acc01[j];
    frow[(size_t)r1 * HD + c0 + m] = acc10[j];
    frow[(size_t)r1 * HD + c1 + m] = acc11[j];
  }
}

// ---------------------------------------------------------------------------
// el[n][h] = <feat[n][h][:], al[h][:]>, er likewise. One wave32 per (n,h).
__global__ __launch_bounds__(256) void k_elr(const float* __restrict__ feat,
                                             const float* __restrict__ al,
                                             const float* __restrict__ ar,
                                             float* __restrict__ el,
                                             float* __restrict__ er) {
  const int gw   = (blockIdx.x * 256 + threadIdx.x) >> 5;  // global wave id
  const int lane = threadIdx.x & 31;
  const int n = gw >> 2, h = gw & 3;
  if (n >= NN) return;
  const float* f = feat + (size_t)n * HD + h * DD;
  const float* a = al + h * DD;
  const float* b = ar + h * DD;
  const float f0 = f[lane], f1 = f[lane + 32];
  float sl = f0 * a[lane] + f1 * a[lane + 32];
  float sr = f0 * b[lane] + f1 * b[lane + 32];
#pragma unroll
  for (int off = 16; off > 0; off >>= 1) {
    sl += __shfl_xor(sl, off, 32);
    sr += __shfl_xor(sr, off, 32);
  }
  if (lane == 0) { el[n * HH + h] = sl; er[n * HH + h] = sr; }
}

// ---------------------------------------------------------------------------
// Edge pass 1: e = leaky_relu(el[src]+er[dst]); store; segment-max into mbuf[dst].
__global__ __launch_bounds__(256) void k_edge_e(const int* __restrict__ src,
                                                const int* __restrict__ dst,
                                                const float* __restrict__ el,
                                                const float* __restrict__ er,
                                                float* __restrict__ ebuf,
                                                float* __restrict__ mbuf) {
  int i = blockIdx.x * 256 + threadIdx.x;      // (edge, head) flat
  if (i >= EE * HH) return;
  const int e = i >> 2, h = i & 3;
  const int s = src[e], d = dst[e];
  float v = el[s * HH + h] + er[d * HH + h];
  v = (v > 0.0f) ? v : NEG_SLOPE * v;
  ebuf[i] = v;
  atomicMaxF32(&mbuf[d * HH + h], v);
}

// Edge pass 2: ex = exp(e - m[dst]); store; segment-sum into sbuf[dst].
__global__ __launch_bounds__(256) void k_edge_s(const int* __restrict__ dst,
                                                const float* __restrict__ mbuf,
                                                float* __restrict__ ebuf,
                                                float* __restrict__ sbuf) {
  int i = blockIdx.x * 256 + threadIdx.x;
  if (i >= EE * HH) return;
  const int e = i >> 2, h = i & 3;
  const int d = dst[e];
  const float ex = __expf(ebuf[i] - mbuf[d * HH + h]);
  ebuf[i] = ex;
  atomicAdd(&sbuf[d * HH + h], ex);
}

// Edge pass 3: out[dst][c] += (ex/max(s,1e-9)) * feat[src][c].
// Block = 256 thr (one per channel c; head = c/64), 8 edges per block.
// feat & out live in L2 (102MB each < 192MB): gathers/atomics are L2-resident.
// Next edge's feat row is prefetched (global_prefetch_b8) to overlap the
// random gather latency with the current edge's atomic issue.
__global__ __launch_bounds__(256) void k_edge_agg(const int* __restrict__ src,
                                                  const int* __restrict__ dst,
                                                  const float* __restrict__ ebuf,
                                                  const float* __restrict__ sbuf,
                                                  const float* __restrict__ feat,
                                                  float* __restrict__ out) {
  const int c = threadIdx.x;
  const int h = c >> 6;
  const int ebase = blockIdx.x * 8;            // EE divisible by 8
  int s = src[ebase];
#pragma unroll
  for (int i = 0; i < 8; ++i) {
    const int e = ebase + i;
    const int d = dst[e];
    const int s_next = (i < 7) ? src[e + 1] : s;
    if (i < 7) {
      __builtin_prefetch(feat + (size_t)s_next * HD + c, 0, 1); // global_prefetch_b8
    }
    const float alpha = ebuf[e * HH + h] / fmaxf(sbuf[d * HH + h], 1e-9f);
    const float f = feat[(size_t)s * HD + c];
    atomicAdd(&out[(size_t)d * HD + c], alpha * f);
    s = s_next;
  }
}

// ---------------------------------------------------------------------------
extern "C" void kernel_launch(void* const* d_in, const int* in_sizes, int n_in,
                              void* d_out, int out_size, void* d_ws, size_t ws_size,
                              hipStream_t stream) {
  const float* x = (const float*)d_in[0];
  float* out = (float*)d_out;

  // Workspace layout (bytes, all 256-aligned): ~142.5 MB total
  char* ws = (char*)d_ws;
  __bf16* xb   = (__bf16*)(ws);                                   // 25,600,000
  __bf16* wb   = (__bf16*)(ws + 25600000);                        //     65,536
  float*  feat = (float*)(ws + 25665536);                         // 102,400,000
  float*  el   = (float*)(ws + 128065536);                        //  1,600,000
  float*  er   = (float*)(ws + 129665536);                        //  1,600,000
  float*  ebuf = (float*)(ws + 131265536);                        //  8,000,000
  float*  mbuf = (float*)(ws + 139265536);                        //  1,600,000
  float*  sbuf = (float*)(ws + 140865536);                        //  1,600,000

  // x -> bf16 once
  k_cvt_x<<<dim3((NN * INF_) / (256 * 4)), dim3(256), 0, stream>>>(x, xb);

  // out = b0 + b1 + b2 (broadcast over nodes)
  k_init_out<<<dim3(NN), dim3(256), 0, stream>>>(
      (const float*)d_in[4], (const float*)d_in[10], (const float*)d_in[16], out);

  const int ehBlocks = (EE * HH + 255) / 256;
  const int msBlocks = (NN * HH + 255) / 256;

  for (int r = 0; r < 3; ++r) {
    const float* W  = (const float*)d_in[1 + r * 6];
    const float* al = (const float*)d_in[2 + r * 6];
    const float* ar = (const float*)d_in[3 + r * 6];
    const int* srcI = (const int*)d_in[5 + r * 6];
    const int* dstI = (const int*)d_in[6 + r * 6];

    // W -> bf16 transposed [256][128]
    k_cvt_w<<<dim3(INF_), dim3(256), 0, stream>>>(W, wb);

    // feat = x @ W  (WMMA bf16 -> fp32), 32-row slabs
    k_gemm_wmma<<<dim3(NN / 32), dim3(256), 0, stream>>>(xb, wb, feat);

    // el / er attention projections
    k_elr<<<dim3((NN * HH) / 8), dim3(256), 0, stream>>>(feat, al, ar, el, er);

    // segment softmax state
    k_init_ms<<<dim3(msBlocks), dim3(256), 0, stream>>>(mbuf, sbuf);
    k_edge_e<<<dim3(ehBlocks), dim3(256), 0, stream>>>(srcI, dstI, el, er, ebuf, mbuf);
    k_edge_s<<<dim3(ehBlocks), dim3(256), 0, stream>>>(dstI, mbuf, ebuf, sbuf);

    // out[dst] += alpha * feat[src]
    k_edge_agg<<<dim3(EE / 8), dim3(256), 0, stream>>>(srcI, dstI, ebuf, sbuf, feat, out);
  }
}
